// ChannelEffFormer_40235253629584
// MI455X (gfx1250) — matrix operand order repaired and checked by reading, not compiled, not run on hardware
//
#include <hip/hip_runtime.h>

// ---------------------------------------------------------------------------
// Types for CDNA5 WMMA (wave32, 16x16x32 bf16 -> f32)
// ---------------------------------------------------------------------------
typedef __attribute__((ext_vector_type(16))) __bf16          v16bf;
typedef __attribute__((ext_vector_type(8)))  float           v8f;
typedef __attribute__((ext_vector_type(8)))  unsigned short  u16x8;
typedef __attribute__((ext_vector_type(16))) unsigned short  u16x16;

__device__ __forceinline__ unsigned short f2bf(float f) {
    unsigned int u = __float_as_uint(f);
    u += 0x7FFFu + ((u >> 16) & 1u);            // round-to-nearest-even
    return (unsigned short)(u >> 16);
}
__device__ __forceinline__ float bf2f(unsigned short h) {
    return __uint_as_float(((unsigned int)h) << 16);
}
__device__ __forceinline__ float gelu_exact(float x) {
    return 0.5f * x * (1.0f + erff(x * 0.70710678118654752f));
}

// ---------------------------------------------------------------------------
// Tiled bf16 WMMA GEMM:  C[M,Nn] = A[M,Kd] * B[Kd,Nn] (+bias) (+residual)
//  - A bf16 row-major (or, with TRANSA, A[m,k] = Aglob[k*lda + m])
//  - B bf16 row-major [Kd, Nn];  Nn must be a multiple of 64
//  - C fp32 or bf16, row-major ldc
//  - batched via blockIdx.z with element strides
// Block: 256 threads = 8 waves, 4(M) x 2(N); tile 64x64, K-step 32.
// Each wave computes a 16x32 strip = two v_wmma per K-step (A frag reused).
// M and K edges use clamped loads + select (branch-free), WMMA full-EXEC.
// ---------------------------------------------------------------------------
#define GEMM_MT 64
#define GEMM_NT 64
#define GEMM_KT 32

template <bool TRANSA, bool RES, bool OUTBF16>
__global__ __launch_bounds__(256)
void gemm_bf16_wmma(const unsigned short* __restrict__ Ag,
                    const unsigned short* __restrict__ Bg,
                    void* __restrict__ Cg,
                    const float* __restrict__ bias,
                    const float* __restrict__ Resid,
                    int M, int Nn, int Kd,
                    int lda, int ldb, int ldc,
                    long strideA, long strideB, long strideC)
{
    __shared__ __align__(16) unsigned short lA[GEMM_MT * GEMM_KT]; // row-major, stride 32
    __shared__ __align__(16) unsigned short lB[GEMM_NT * GEMM_KT]; // col-major: lB[n*32+k]

    const int t     = threadIdx.x;
    const int lane  = t & 31;
    const int wave  = t >> 5;
    const int wm    = wave >> 1;   // 0..3
    const int wn    = wave & 1;    // 0..1
    const int bz    = blockIdx.z;
    const unsigned short* A = Ag + (long)bz * strideA;
    const unsigned short* B = Bg + (long)bz * strideB;
    const int blockM = blockIdx.y * GEMM_MT;
    const int blockN = blockIdx.x * GEMM_NT;

    const int mrow  = lane & 15;   // row (A) or col (B/C) within 16
    const int khalf = lane >> 4;   // lane-group selector

    // A-tile mapping: 8 contiguous K per thread
    const int a_r  = (t * 8) >> 5;          // tile row 0..63
    const int a_c  = (t * 8) & 31;          // tile col 0..24
    // B-tile mapping: 8 contiguous N per thread
    const int b_k  = t >> 3;                // 0..31
    const int b_n0 = (t & 7) * 8;           // 0..56

    v8f acc0 = {};
    v8f acc1 = {};

    for (int k0 = 0; k0 < Kd; k0 += GEMM_KT) {
        const bool kFull = (k0 + GEMM_KT) <= Kd;       // block-uniform
        // ---- stage A tile (64x32) ----
        {
            const int gm = blockM + a_r;
            if (!TRANSA && kFull && (blockM + GEMM_MT) <= M) {
                // interior fast path: one b128 load per thread
                *(u16x8*)&lA[t * 8] =
                    *(const u16x8*)(A + (long)gm * lda + (k0 + a_c));
            } else {
                // edge path: clamped unconditional loads + select (no branches)
                const int gmc = (gm < M) ? gm : (M - 1);
                #pragma unroll
                for (int j = 0; j < 8; ++j) {
                    int kk  = k0 + a_c + j;
                    int kkc = (kk < Kd) ? kk : (Kd - 1);
                    unsigned short v = TRANSA ? A[(long)kkc * lda + gmc]
                                              : A[(long)gmc * lda + kkc];
                    lA[t * 8 + j] = (gm < M && kk < Kd) ? v : (unsigned short)0;
                }
            }
        }
        // ---- stage B tile (32K x 64N), transposed to column-major in LDS ----
        {
            const int gk  = k0 + b_k;
            const int gkc = (gk < Kd) ? gk : (Kd - 1);
            const u16x8 vv =
                *(const u16x8*)(B + (long)gkc * ldb + (blockN + b_n0));
            #pragma unroll
            for (int j = 0; j < 8; ++j) {
                unsigned short v = (gk < Kd) ? vv[j] : (unsigned short)0;
                lB[(b_n0 + j) * GEMM_KT + b_k] = v;
            }
        }
        // prefetch next K tiles (speculative)
        if (kFull && (k0 + GEMM_KT) < Kd) {
            if (!TRANSA) {
                int gm = blockM + a_r; if (gm >= M) gm = M - 1;
                __builtin_prefetch(A + (long)gm * lda + (k0 + GEMM_KT + a_c), 0, 1);
            }
            __builtin_prefetch(B + (long)(k0 + GEMM_KT + b_k) * ldb + blockN + b_n0, 0, 1);
        }
        __syncthreads();

        // ---- build fragments (16-bit A 16x32 / B 32x16, contiguous k = 16*khalf + s in both) ----
        // A lane L: m = L%16; elems = 16 consecutive K starting at 16*khalf
        const unsigned short* arow = &lA[(wm * 16 + mrow) * GEMM_KT];
        u16x16 au;
        #pragma unroll
        for (int j = 0; j < 8; ++j) {
            au[j]     = arow[16 * khalf + j];
            au[8 + j] = arow[16 * khalf + 8 + j];
        }
        // B lane L: n = L%16; elems = 16 consecutive K starting at 16*khalf
        const unsigned short* bcol0 = &lB[(wn * 32 + mrow) * GEMM_KT];
        const unsigned short* bcol1 = &lB[(wn * 32 + 16 + mrow) * GEMM_KT];
        u16x16 bu0, bu1;
        #pragma unroll
        for (int j = 0; j < 16; ++j) {
            bu0[j] = bcol0[16 * khalf + j];
            bu1[j] = bcol1[16 * khalf + j];
        }

        v16bf af  = __builtin_bit_cast(v16bf, au);
        v16bf bf0 = __builtin_bit_cast(v16bf, bu0);
        v16bf bf1 = __builtin_bit_cast(v16bf, bu1);
        acc0 = __builtin_amdgcn_wmma_f32_16x16x32_bf16(
                 false, af, false, bf0, (short)0, acc0, false, false);
        acc1 = __builtin_amdgcn_wmma_f32_16x16x32_bf16(
                 false, af, false, bf1, (short)0, acc1, false, false);
        __syncthreads();
    }

    // ---- epilogue: C/D layout lanes0-15 -> M=r, lanes16-31 -> M=8+r ----
    const int col0 = blockN + wn * 32 + mrow;
    const int col1 = col0 + 16;
    const int row0 = blockM + wm * 16 + 8 * khalf;
    const float bv0 = bias ? bias[col0] : 0.0f;
    const float bv1 = bias ? bias[col1] : 0.0f;
    #pragma unroll
    for (int r = 0; r < 8; ++r) {
        int row = row0 + r;
        if (row < M) {
            long off0 = (long)bz * strideC + (long)row * ldc + col0;
            long off1 = off0 + 16;
            float v0 = acc0[r] + bv0;
            float v1 = acc1[r] + bv1;
            if (RES) { v0 += Resid[off0]; v1 += Resid[off1]; }
            if (OUTBF16) {
                ((unsigned short*)Cg)[off0] = f2bf(v0);
                ((unsigned short*)Cg)[off1] = f2bf(v1);
            } else {
                ((float*)Cg)[off0] = v0;
                ((float*)Cg)[off1] = v1;
            }
        }
    }
}

// ---------------------------------------------------------------------------
// LayerNorm: one wave per row (wave32); optional fused exact GELU; fp32/bf16 out
// ---------------------------------------------------------------------------
template <bool GELU, bool OUTBF16>
__global__ void layernorm_k(const float* __restrict__ in,
                            const float* __restrict__ g,
                            const float* __restrict__ b,
                            void* __restrict__ out, int rows, int D)
{
    int row = blockIdx.x * (blockDim.x >> 5) + (threadIdx.x >> 5);
    if (row >= rows) return;
    int lane = threadIdx.x & 31;
    const float* x = in + (long)row * D;
    float s = 0.f, s2 = 0.f;
    for (int i = lane; i < D; i += 32) { float v = x[i]; s += v; s2 += v * v; }
    for (int o = 16; o > 0; o >>= 1) { s += __shfl_xor(s, o, 32); s2 += __shfl_xor(s2, o, 32); }
    float mean = s / D;
    float var  = s2 / D - mean * mean;          // biased variance (torch LN)
    float rstd = rsqrtf(var + 1e-5f);
    for (int i = lane; i < D; i += 32) {
        float v = (x[i] - mean) * rstd * g[i] + b[i];
        if (GELU) v = gelu_exact(v);
        long off = (long)row * D + i;
        if (OUTBF16) ((unsigned short*)out)[off] = f2bf(v);
        else         ((float*)out)[off] = v;
    }
}

// softmax over channels (axis=2): wave per row, bf16 out
__global__ void softmax_row_k(const float* __restrict__ in,
                              unsigned short* __restrict__ out, int rows, int D)
{
    int row = blockIdx.x * (blockDim.x >> 5) + (threadIdx.x >> 5);
    if (row >= rows) return;
    int lane = threadIdx.x & 31;
    const float* x = in + (long)row * D;
    float mx = -3.4e38f;
    for (int i = lane; i < D; i += 32) mx = fmaxf(mx, x[i]);
    for (int o = 16; o > 0; o >>= 1) mx = fmaxf(mx, __shfl_xor(mx, o, 32));
    float s = 0.f;
    for (int i = lane; i < D; i += 32) s += __expf(x[i] - mx);
    for (int o = 16; o > 0; o >>= 1) s += __shfl_xor(s, o, 32);
    float inv = 1.0f / s;
    for (int i = lane; i < D; i += 32)
        out[(long)row * D + i] = f2bf(__expf(x[i] - mx) * inv);
}

// softmax over tokens (axis=1): wave per (batch, channel) column, bf16 out
__global__ void softmax_col_k(const float* __restrict__ in,
                              unsigned short* __restrict__ out,
                              int Bn, int Ntok, int D)
{
    int col = blockIdx.x * (blockDim.x >> 5) + (threadIdx.x >> 5);
    if (col >= Bn * D) return;
    int lane = threadIdx.x & 31;
    int b = col / D, c = col % D;
    const float* base = in + (long)b * Ntok * D + c;
    float mx = -3.4e38f;
    for (int i = lane; i < Ntok; i += 32) mx = fmaxf(mx, base[(long)i * D]);
    for (int o = 16; o > 0; o >>= 1) mx = fmaxf(mx, __shfl_xor(mx, o, 32));
    float s = 0.f;
    for (int i = lane; i < Ntok; i += 32) s += __expf(base[(long)i * D] - mx);
    for (int o = 16; o > 0; o >>= 1) s += __shfl_xor(s, o, 32);
    float inv = 1.0f / s;
    unsigned short* ob = out + (long)b * Ntok * D + c;
    for (int i = lane; i < Ntok; i += 32)
        ob[(long)i * D] = f2bf(__expf(base[(long)i * D] - mx) * inv);
}

// 3x3 depthwise conv + bias + skip, tokens layout [B, H*W, C] fp32
__global__ void dwconv_skip_k(const float* __restrict__ a,
                              const float* __restrict__ w,
                              const float* __restrict__ bias,
                              float* __restrict__ out,
                              int Bn, int H, int W, int C)
{
    long idx = (long)blockIdx.x * blockDim.x + threadIdx.x;
    long total = (long)Bn * H * W * C;
    if (idx >= total) return;
    int c = (int)(idx % C);
    long t2 = idx / C;
    int xw = (int)(t2 % W);
    long t3 = t2 / W;
    int yh = (int)(t3 % H);
    int b  = (int)(t3 / H);
    float s = bias[c];
    #pragma unroll
    for (int ky = 0; ky < 3; ++ky) {
        int iy = yh + ky - 1;
        if (iy < 0 || iy >= H) continue;
        #pragma unroll
        for (int kx = 0; kx < 3; ++kx) {
            int ix = xw + kx - 1;
            if (ix < 0 || ix >= W) continue;
            s += a[((long)(b * H + iy) * W + ix) * C + c] * w[c * 9 + ky * 3 + kx];
        }
    }
    out[idx] = s + a[idx];   // skip around dwconv
}

// strided conv patch embed: NCHW in -> tokens [B, OH*OW, Co] fp32
__global__ void conv_pe_k(const float* __restrict__ in, const float* __restrict__ w,
                          const float* __restrict__ bias, float* __restrict__ out,
                          int Bn, int Cin, int H, int W, int Co,
                          int K, int S, int Pd, int OH, int OW)
{
    long idx = (long)blockIdx.x * blockDim.x + threadIdx.x;
    long total = (long)Bn * OH * OW * Co;
    if (idx >= total) return;
    int oc = (int)(idx % Co);
    long t = idx / Co;
    int n = (int)(t % (OH * OW));
    int b = (int)(t / (OH * OW));
    int oy = n / OW, ox = n % OW;
    float s = bias[oc];
    for (int ci = 0; ci < Cin; ++ci) {
        const float* ip = in + (long)(b * Cin + ci) * H * W;
        const float* wp = w + (long)(oc * Cin + ci) * K * K;
        for (int ky = 0; ky < K; ++ky) {
            int iy = oy * S - Pd + ky;
            if (iy < 0 || iy >= H) continue;
            for (int kx = 0; kx < K; ++kx) {
                int ix = ox * S - Pd + kx;
                if (ix < 0 || ix >= W) continue;
                s += ip[(long)iy * W + ix] * wp[ky * K + kx];
            }
        }
    }
    out[idx] = s;
}

// fp32 -> bf16 elementwise
__global__ void cast_bf16_k(const float* __restrict__ in,
                            unsigned short* __restrict__ out, long n)
{
    long i = (long)blockIdx.x * blockDim.x + threadIdx.x;
    if (i < n) out[i] = f2bf(in[i]);
}

// transpose [Nout,Kin] fp32 -> [Kin,Nout] bf16  (proj weights [out,in] -> B[k,n])
__global__ void packT_bf16_k(const float* __restrict__ in,
                             unsigned short* __restrict__ out, int Nout, int Kin)
{
    long i = (long)blockIdx.x * blockDim.x + threadIdx.x;
    if (i >= (long)Nout * Kin) return;
    int n = (int)(i / Kin);
    int k = (int)(i % Kin);
    out[(long)k * Nout + n] = f2bf(in[i]);
}

// tokens [B,N,C] fp32 -> NCHW [B,C,N]
__global__ void tok2nchw_k(const float* __restrict__ in, float* __restrict__ out,
                           int Bn, int Ntok, int C)
{
    long i = (long)blockIdx.x * blockDim.x + threadIdx.x;
    long total = (long)Bn * Ntok * C;
    if (i >= total) return;
    int c = (int)(i % C);
    long t = i / C;
    int n = (int)(t % Ntok);
    int b = (int)(t / Ntok);
    out[((long)b * C + c) * Ntok + n] = in[i];
}

// ---------------------------------------------------------------------------
// Host orchestration
// ---------------------------------------------------------------------------
extern "C" void kernel_launch(void* const* d_in, const int* in_sizes, int n_in,
                              void* d_out, int out_size, void* d_ws, size_t ws_size,
                              hipStream_t stream)
{
    (void)in_sizes; (void)n_in; (void)out_size; (void)ws_size;
    const int Bn = 8;
    const int IN_DIM_[3] = {128, 320, 512};
    const int LAYERS_[3] = {2, 2, 2};
    const int PATCH_[3]  = {7, 3, 3};
    const int STRIDE_[3] = {4, 2, 2};
    const int PAD_[3]    = {3, 1, 1};
    const int IN_CH_[3]  = {3, 128, 320};

    // workspace bump layout (maxima come from stage 1: M=25088, d=128, h=512)
    char* ws = (char*)d_ws;
    float*          bX   = (float*)(ws + 0);                       // residual [M,d] f32
    float*          bA   = (float*)(ws + 12845056u);               // [M,h] f32 scratch
    float*          bB   = (float*)(ws + 64225280u);               // [M,h] f32 scratch
    unsigned short* bC   = (unsigned short*)(ws + 115605504u);     // [M,h] bf16 scratch
    unsigned short* bXN  = (unsigned short*)(ws + 141295616u);     // [M,d] bf16
    unsigned short* bV   = (unsigned short*)(ws + 147718144u);     // [M,d] bf16
    unsigned short* bCTX = (unsigned short*)(ws + 154140672u);     // [B,d,d] bf16
    unsigned short* bW   = (unsigned short*)(ws + 158334976u);     // weight bf16

    // ---- parameter walk (JAX pytree order: dict keys sorted) ----
    struct MlpP { const float *dw_b,*dw_w,*fc1_b,*fc1_w,*fc2_b,*fc2_w,*n_b,*n_g; };
    struct BlkP { const float *k_b,*k_w,*ln1_b,*ln1_g,*ln2_b,*ln2_g,*ln3_b,*ln3_g;
                  MlpP mlp1, mlp2;
                  const float *q_b,*q_w,*r_b,*r_w,*v_b,*v_w; };
    struct StageP { BlkP blk[2];
                    const float *norm_b,*norm_g,*pe_b,*pe_ln_b,*pe_ln_g,*pe_w; };

    int cur = 1;   // d_in[0] = x
    auto P = [&]() -> const float* { return (const float*)d_in[cur++]; };

    StageP st[3];
    for (int s = 0; s < 3; ++s) {
        for (int l = 0; l < LAYERS_[s]; ++l) {
            BlkP& bk = st[s].blk[l];
            bk.k_b = P(); bk.k_w = P();
            bk.ln1_b = P(); bk.ln1_g = P();
            bk.ln2_b = P(); bk.ln2_g = P();
            bk.ln3_b = P(); bk.ln3_g = P();
            MlpP* ms[2] = { &bk.mlp1, &bk.mlp2 };
            for (int j = 0; j < 2; ++j) {
                MlpP& m = *ms[j];
                m.dw_b = P(); m.dw_w = P();
                m.fc1_b = P(); m.fc1_w = P();
                m.fc2_b = P(); m.fc2_w = P();
                m.n_b = P(); m.n_g = P();
            }
            bk.q_b = P(); bk.q_w = P();
            bk.r_b = P(); bk.r_w = P();
            bk.v_b = P(); bk.v_w = P();
        }
        st[s].norm_b = P(); st[s].norm_g = P();
        st[s].pe_b = P(); st[s].pe_ln_b = P(); st[s].pe_ln_g = P(); st[s].pe_w = P();
    }

    size_t outOff = 0;
    int H = 224, W = 224;
    const float* curIn = (const float*)d_in[0];

    for (int s = 0; s < 3; ++s) {
        const int d = IN_DIM_[s], cin = IN_CH_[s];
        const int kk = PATCH_[s], sd = STRIDE_[s], pd = PAD_[s];
        const int OH = (H + 2 * pd - kk) / sd + 1;
        const int OW = (W + 2 * pd - kk) / sd + 1;
        const int N = OH * OW, M = Bn * N, h = 4 * d;

        // patch embed -> tokens (bA), then LN -> residual stream (bX)
        {
            long tot = (long)M * d;
            conv_pe_k<<<(unsigned)((tot + 255) / 256), 256, 0, stream>>>(
                curIn, st[s].pe_w, st[s].pe_b, bA, Bn, cin, H, W, d, kk, sd, pd, OH, OW);
        }
        layernorm_k<false, false><<<(M + 7) / 8, 256, 0, stream>>>(
            bA, st[s].pe_ln_g, st[s].pe_ln_b, bX, M, d);

        for (int l = 0; l < LAYERS_[s]; ++l) {
            const BlkP& bk = st[s].blk[l];

            // ================= efficient attention =================
            layernorm_k<false, true><<<(M + 7) / 8, 256, 0, stream>>>(
                bX, bk.ln1_g, bk.ln1_b, bXN, M, d);

            long dd = (long)d * d;
            // k_lin (f32)
            packT_bf16_k<<<(unsigned)((dd + 255) / 256), 256, 0, stream>>>(bk.k_w, bW, d, d);
            gemm_bf16_wmma<false, false, false><<<dim3(d / 64, (M + 63) / 64, 1), 256, 0, stream>>>(
                bXN, bW, bA, bk.k_b, nullptr, M, d, d, d, d, d, 0, 0, 0);
            // q_lin (f32)
            packT_bf16_k<<<(unsigned)((dd + 255) / 256), 256, 0, stream>>>(bk.q_w, bW, d, d);
            gemm_bf16_wmma<false, false, false><<<dim3(d / 64, (M + 63) / 64, 1), 256, 0, stream>>>(
                bXN, bW, bB, bk.q_b, nullptr, M, d, d, d, d, d, 0, 0, 0);
            // v (bf16)
            packT_bf16_k<<<(unsigned)((dd + 255) / 256), 256, 0, stream>>>(bk.v_w, bW, d, d);
            gemm_bf16_wmma<false, false, true><<<dim3(d / 64, (M + 63) / 64, 1), 256, 0, stream>>>(
                bXN, bW, bV, bk.v_b, nullptr, M, d, d, d, d, d, 0, 0, 0);

            unsigned short* KS = bC;
            unsigned short* QS = bC + (size_t)M * d;
            softmax_col_k<<<(Bn * d + 7) / 8, 256, 0, stream>>>(bA, KS, Bn, N, d);
            softmax_row_k<<<(M + 7) / 8, 256, 0, stream>>>(bB, QS, M, d);

            // ctx[b] = KS[b]^T (d x N) * V[b] (N x d)  -> bf16 [B,d,d]
            gemm_bf16_wmma<true, false, true><<<dim3(d / 64, (d + 63) / 64, Bn), 256, 0, stream>>>(
                KS, bV, bCTX, nullptr, nullptr, d, d, N, d, d, d,
                (long)N * d, (long)N * d, (long)d * d);
            // attn[b] = QS[b] (N x d) * ctx[b] (d x d) -> bf16 (reuse bXN)
            gemm_bf16_wmma<false, false, true><<<dim3(d / 64, (N + 63) / 64, Bn), 256, 0, stream>>>(
                QS, bCTX, bXN, nullptr, nullptr, N, d, d, d, d, d,
                (long)N * d, (long)d * d, (long)N * d);
            // r projection + residual -> bX
            packT_bf16_k<<<(unsigned)((dd + 255) / 256), 256, 0, stream>>>(bk.r_w, bW, d, d);
            gemm_bf16_wmma<false, true, false><<<dim3(d / 64, (M + 63) / 64, 1), 256, 0, stream>>>(
                bXN, bW, bX, bk.r_b, bX, M, d, d, d, d, d, 0, 0, 0);

            // ================= two MixFFNs =================
            for (int j = 0; j < 2; ++j) {
                const MlpP& m = (j == 0) ? bk.mlp1 : bk.mlp2;
                const float* lg = (j == 0) ? bk.ln2_g : bk.ln3_g;
                const float* lb = (j == 0) ? bk.ln2_b : bk.ln3_b;

                layernorm_k<false, true><<<(M + 7) / 8, 256, 0, stream>>>(
                    bX, lg, lb, bXN, M, d);
                // fc1: [M,d] x [d,h] -> bA f32
                cast_bf16_k<<<(unsigned)(((long)d * h + 255) / 256), 256, 0, stream>>>(
                    m.fc1_w, bW, (long)d * h);
                gemm_bf16_wmma<false, false, false><<<dim3(h / 64, (M + 63) / 64, 1), 256, 0, stream>>>(
                    bXN, bW, bA, m.fc1_b, nullptr, M, h, d, d, h, h, 0, 0, 0);
                // dwconv + skip -> bB
                {
                    long tot = (long)M * h;
                    dwconv_skip_k<<<(unsigned)((tot + 255) / 256), 256, 0, stream>>>(
                        bA, m.dw_w, m.dw_b, bB, Bn, OH, OW, h);
                }
                // LN + GELU -> bf16 bC
                layernorm_k<true, true><<<(M + 7) / 8, 256, 0, stream>>>(
                    bB, m.n_g, m.n_b, bC, M, h);
                // fc2: [M,h] x [h,d] + residual -> bX
                cast_bf16_k<<<(unsigned)(((long)h * d + 255) / 256), 256, 0, stream>>>(
                    m.fc2_w, bW, (long)h * d);
                gemm_bf16_wmma<false, true, false><<<dim3(d / 64, (M + 63) / 64, 1), 256, 0, stream>>>(
                    bC, bW, bX, m.fc2_b, bX, M, d, h, h, d, d, 0, 0, 0);
            }
        }

        // final stage LN -> bA, transpose tokens -> NCHW into d_out
        layernorm_k<false, false><<<(M + 7) / 8, 256, 0, stream>>>(
            bX, st[s].norm_g, st[s].norm_b, bA, M, d);
        {
            long tot = (long)M * d;
            tok2nchw_k<<<(unsigned)((tot + 255) / 256), 256, 0, stream>>>(
                bA, (float*)d_out + outOff, Bn, N, d);
        }
        curIn = (const float*)d_out + outOff;   // next stage reads our NCHW output
        outOff += (size_t)M * d;
        H = OH; W = OW;
    }
}
